// VQVAE_45483703665096
// MI455X (gfx1250) — matrix operand order repaired and checked by reading, not compiled, not run on hardware
//
#include <hip/hip_runtime.h>
#include <hip/hip_bf16.h>

// ---------------- CDNA5 WMMA / TDM types ----------------
typedef __attribute__((ext_vector_type(16))) __bf16          v16bf;
typedef __attribute__((ext_vector_type(8)))  float           v8f;
typedef __attribute__((ext_vector_type(4)))  unsigned int    v4u;
typedef __attribute__((ext_vector_type(8)))  int             v8i;
typedef __attribute__((ext_vector_type(4)))  int             v4i;

// =====================================================================
// Generic implicit-GEMM conv / transposed-conv kernel, fully templated.
// Block = 128 threads (4 wave32), tile = 64(M) x 64(N), K-chunk = 32.
// Each wave: 1 B-fragment + 4 A-fragments -> 4 x v_wmma_f32_16x16x32_bf16.
// OP: 0 none, 1 relu, 2 sigmoid.
// =====================================================================
template<int CIN, int IH, int IW, int COUT, int OH, int OW,
         int KS, int STRIDE, int EFFPAD, int UPDIL, int WLAYOUT, int OP>
__global__ __launch_bounds__(128) void k_conv_wmma(
        const float* __restrict__ in, const float* __restrict__ wgt,
        const float* __restrict__ bias, float* __restrict__ out) {
    constexpr int BATCH = 128;
    constexpr int KK2 = KS * KS;
    constexpr int KC  = CIN * KK2;
    constexpr int OHW = OH * OW;
    constexpr int M   = BATCH * OHW;

    __shared__ __bf16 As[64][34];   // 64 x 32 bf16, padded rows (17-bank stride)
    __shared__ __bf16 Bs[32][64];   // 32 x 64 bf16

    const int t    = threadIdx.x;
    const int wave = t >> 5, lane = t & 31;
    const int half = lane >> 4, l16 = lane & 15;
    const int m0   = blockIdx.x * 64;
    const int n0   = blockIdx.y * 64;

    __builtin_prefetch(wgt, 0, 3);          // global_prefetch_b8

    v8f acc[4] = {v8f{}, v8f{}, v8f{}, v8f{}};

    #pragma unroll 1
    for (int kc0 = 0; kc0 < KC; kc0 += 32) {
        // ---- stage A: 64x32 im2col gather, 16 elements/thread ----
        #pragma unroll
        for (int i = 0; i < 16; ++i) {
            int e = t * 16 + i;
            int row = e >> 5, kl = e & 31;
            int kg  = kc0 + kl;
            int m   = m0 + row;
            float v = 0.0f;
            if ((KC % 32 == 0 || kg < KC) && m < M) {
                int ci = kg / KK2, r = kg % KK2;
                int ky = r / KS, kx = r % KS;
                int b  = m / OHW, r2 = m % OHW;
                int oy = r2 / OW, ox = r2 % OW;
                int iy = oy * STRIDE - EFFPAD + ky;
                int ix = ox * STRIDE - EFFPAD + kx;
                bool ok = (iy >= 0) && (ix >= 0);
                int uy = iy, ux = ix;
                if (UPDIL > 1) {
                    ok = ok && ((iy % UPDIL) == 0) && ((ix % UPDIL) == 0);
                    uy = iy / UPDIL; ux = ix / UPDIL;
                }
                if (ok && uy < IH && ux < IW)
                    v = in[((b * CIN + ci) * IH + uy) * IW + ux];
            }
            As[row][kl] = (__bf16)v;        // native v_cvt f32->bf16
        }
        // ---- stage B: 32x64 weights, 16 elements/thread ----
        #pragma unroll
        for (int i = 0; i < 16; ++i) {
            int kl, nl;
            if (WLAYOUT == 0) { int e = t * 16 + i; kl = e & 31; nl = e >> 5; }
            else              { int e = t + i * 128; kl = e >> 6; nl = e & 63; }
            int kg = kc0 + kl, n = n0 + nl;
            float v = 0.0f;
            if ((KC % 32 == 0 || kg < KC) && (COUT % 64 == 0 || n < COUT)) {
                if (WLAYOUT == 0) {
                    v = wgt[n * KC + kg];                       // OIHW flat
                } else {                                        // (Cin,Cout,kh,kw), flipped
                    int ci = kg / KK2, r = kg % KK2;
                    v = wgt[ci * (COUT * KK2) + n * KK2 + (KK2 - 1 - r)];
                }
            }
            Bs[kl][nl] = (__bf16)v;
        }
        __syncthreads();

        // ---- fragments (CDNA5 16-bit WMMA layouts) + 4 WMMAs ----
        v16bf bu;
        const int nl = wave * 16 + l16;
        #pragma unroll
        for (int j = 0; j < 8; ++j) {
            int kb = (j < 4 ? 2 * j : 16 + 2 * (j - 4)) + 8 * half;
            bu[2 * j] = Bs[kb][nl]; bu[2 * j + 1] = Bs[kb + 1][nl];
        }
        #pragma unroll
        for (int s = 0; s < 4; ++s) {
            v16bf au;
            const int mr = s * 16 + l16;
            #pragma unroll
            for (int j = 0; j < 4; ++j) {
                int kb = 2 * j + 8 * half;
                au[2 * j] = As[mr][kb]; au[2 * j + 1] = As[mr][kb + 1];
                int kb2 = 16 + 2 * j + 8 * half;
                au[8 + 2 * j] = As[mr][kb2]; au[8 + 2 * j + 1] = As[mr][kb2 + 1];
            }
            acc[s] = __builtin_amdgcn_wmma_f32_16x16x32_bf16(
                         false, au, false, bu, (short)0, acc[s], false, false);
        }
        __syncthreads();
    }

    // ---- epilogue: bias + activation, NCHW store ----
    const int n = n0 + wave * 16 + l16;
    if (COUT % 64 != 0 && n >= COUT) return;
    const float bn = bias[n];
    #pragma unroll
    for (int s = 0; s < 4; ++s) {
        #pragma unroll
        for (int j = 0; j < 8; ++j) {
            int m = m0 + s * 16 + j + 8 * half;
            if (M % 64 != 0 && m >= M) continue;
            float v = acc[s][j] + bn;
            if (OP == 1)      v = fmaxf(v, 0.0f);
            else if (OP == 2) v = 1.0f / (1.0f + __expf(-v));
            int b  = m / OHW, r2 = m % OHW;
            int oy = r2 / OW, ox = r2 % OW;
            out[((b * COUT + n) * OH + oy) * OW + ox] = v;
        }
    }
}

// =====================================================================
// VQ distance GEMM: dist[m][n] = ||e_n||^2 - 2 * flat[m].e_n
// M=32768, N=512, K=64.  64x64 block tile, fp32 tiles staged by the
// Tensor Data Mover (TDM) when available, bf16-converted at frag build.
// LDS tiles use TDM padding (2 DWORDs every 64) -> row stride 66 floats.
// =====================================================================
#if __has_builtin(__builtin_amdgcn_tensor_load_to_lds)
#define VQ_HAS_TDM 1
#else
#define VQ_HAS_TDM 0
#endif

#if VQ_HAS_TDM
__device__ __forceinline__ void tdm_load_16kb(unsigned lds_byte_off, const float* gsrc) {
    unsigned long long ga = (unsigned long long)(const void*)gsrc;
    v4u g0;
    g0[0] = 1u;                                   // count=1, user descriptor
    g0[1] = lds_byte_off;                         // LDS byte address
    g0[2] = (unsigned)(ga & 0xffffffffu);         // global_addr[31:0]
    g0[3] = (unsigned)((ga >> 32) & 0x1ffffffu)   // global_addr[56:32]
            | 0x80000000u;                        // type=2 ("image") at [127:126]
    v8i g1;
    g1[0] = (2 << 16)                             // data_size = 4 bytes
          | (1 << 20)                             // pad_enable
          | (5 << 22)                             // pad_interval: every 64 DWORDs
          | (1 << 25);                            // pad_amount: 2 DWORDs
    g1[1] = (int)(4096u << 16);                   // tensor_dim0[15:0]=4096 (atomic addr 0)
    g1[2] = 0;                                    // tensor_dim0 hi=0, tensor_dim1 lo=0
    g1[3] = (int)(4096u << 16);                   // tile_dim0 = 4096 (1-D tile)
    g1[4] = 0;                                    // tile_dim1=0 (unused), tile_dim2=0
    g1[5] = 4096;                                 // tensor_dim0_stride (unused for 1-D)
    g1[6] = 0;
    g1[7] = 0;
    v4i z4 = {0, 0, 0, 0};                        // D# groups 2/3: unused (<=2-D)
    v8i z8 = {0, 0, 0, 0, 0, 0, 0, 0};            // trailing group (6-arg toolchain form)
    __builtin_amdgcn_tensor_load_to_lds(g0, g1, z4, z4, z8, 0);
}
__device__ __forceinline__ void tdm_wait0() {
#if __has_builtin(__builtin_amdgcn_s_wait_tensorcnt)
    __builtin_amdgcn_s_wait_tensorcnt(0);
#else
    asm volatile("s_wait_tensorcnt 0x0" ::: "memory");
#endif
}
#endif

__global__ __launch_bounds__(128) void k_vq_dist_wmma(
        const float* __restrict__ z, const float* __restrict__ cb,
        const float* __restrict__ cnorm, float* __restrict__ dist) {
    __shared__ __align__(16) float Asf[64 * 66];  // 64 rows, stride 66 (TDM pad)
    __shared__ __align__(16) float Bsf[64 * 66];

    const int t    = threadIdx.x;
    const int wave = t >> 5, lane = t & 31;
    const int half = lane >> 4, l16 = lane & 15;
    const long m0  = (long)blockIdx.x * 64;
    const int  n0  = blockIdx.y * 64;

#if VQ_HAS_TDM
    if (t < 32) {                                  // wave 0 drives both DMAs
        tdm_load_16kb((unsigned)(uintptr_t)(void*)Asf, z  + m0 * 64);
        tdm_load_16kb((unsigned)(uintptr_t)(void*)Bsf, cb + (long)n0 * 64);
        tdm_wait0();
    }
#else
    #pragma unroll
    for (int i = 0; i < 32; ++i) {                 // manual fallback staging
        int e = t * 32 + i, r = e >> 6, c = e & 63;
        Asf[r * 66 + c] = z[m0 * 64 + e];
        Bsf[r * 66 + c] = cb[(long)n0 * 64 + e];
    }
#endif
    __syncthreads();

    v8f acc[4] = {v8f{}, v8f{}, v8f{}, v8f{}};
    const int nl = wave * 16 + l16;

    #pragma unroll
    for (int kc = 0; kc < 2; ++kc) {
        v16bf bu;                                  // B[kb][n] = codebook[n][kb]
        #pragma unroll
        for (int j = 0; j < 8; ++j) {
            int kb = kc * 32 + (j < 4 ? 2 * j : 16 + 2 * (j - 4)) + 8 * half;
            float2 f = *(const float2*)&Bsf[nl * 66 + kb];
            bu[2 * j] = (__bf16)f.x; bu[2 * j + 1] = (__bf16)f.y;
        }
        #pragma unroll
        for (int s = 0; s < 4; ++s) {
            v16bf au;
            const int mr = s * 16 + l16;
            #pragma unroll
            for (int j = 0; j < 4; ++j) {
                int kb = kc * 32 + 2 * j + 8 * half;
                float2 f = *(const float2*)&Asf[mr * 66 + kb];
                au[2 * j] = (__bf16)f.x; au[2 * j + 1] = (__bf16)f.y;
                int kb2 = kc * 32 + 16 + 2 * j + 8 * half;
                float2 g = *(const float2*)&Asf[mr * 66 + kb2];
                au[8 + 2 * j] = (__bf16)g.x; au[8 + 2 * j + 1] = (__bf16)g.y;
            }
            acc[s] = __builtin_amdgcn_wmma_f32_16x16x32_bf16(
                         false, au, false, bu, (short)0, acc[s], false, false);
        }
    }

    const int n = n0 + wave * 16 + l16;
    const float cn = cnorm[n];
    #pragma unroll
    for (int s = 0; s < 4; ++s)
        #pragma unroll
        for (int j = 0; j < 8; ++j) {
            long m = m0 + s * 16 + j + 8 * half;
            dist[m * 512 + n] = cn - 2.0f * acc[s][j];
        }
}

// ---------------- codebook squared norms ----------------
__global__ void k_cnorm(const float* cb, float* cnorm, int K, int D) {
    int j = blockIdx.x * blockDim.x + threadIdx.x;
    if (j >= K) return;
    float s = 0.0f;
    for (int d = 0; d < D; ++d) { float v = cb[j * D + d]; s += v * v; }
    cnorm[j] = s;
}

__global__ void k_init(float* counts, float* sumsq, int K) {
    int i = blockIdx.x * blockDim.x + threadIdx.x;
    if (i < K) counts[i] = 0.0f;
    if (i == 0) *sumsq = 0.0f;
}

// ---------------- per-row argmin + gather + stats (one wave32 per row) ------
__global__ __launch_bounds__(256) void k_vq_argmin(
    const float* dist, const float* z, const float* codebook,
    float* qout, float* counts, float* sumsq, int N, int K, int D) {
    int wave = threadIdx.x >> 5, lane = threadIdx.x & 31;
    int row  = blockIdx.x * 8 + wave;
    if (row >= N) return;
    const float* dr = dist + (long)row * K;
    float best = 3.4e38f; int bidx = 0x7fffffff;
    for (int j = lane; j < K; j += 32) {
        float v = dr[j];
        if (v < best) { best = v; bidx = j; }
    }
    #pragma unroll
    for (int off = 16; off > 0; off >>= 1) {       // wave32 butterfly argmin
        float ob = __shfl_xor(best, off, 32);
        int   oi = __shfl_xor(bidx, off, 32);
        if (ob < best || (ob == best && oi < bidx)) { best = ob; bidx = oi; }
    }
    const float* cbr = codebook + (long)bidx * D;
    const float* zr  = z + (long)row * D;
    float*       qr  = qout + (long)row * D;
    float ss = 0.0f;
    for (int d = lane; d < D; d += 32) {
        float q = cbr[d];
        qr[d] = q;
        float df = q - zr[d];
        ss += df * df;
    }
    #pragma unroll
    for (int off = 16; off > 0; off >>= 1) ss += __shfl_xor(ss, off, 32);
    if (lane == 0) {
        atomicAdd(&counts[bidx], 1.0f);
        atomicAdd(sumsq, ss);
    }
}

// ---------------- scalars: vq_loss and perplexity (one wave32) --------------
__global__ void k_finalize(const float* counts, const float* sumsq,
                           float* out_loss, float* out_perp,
                           int K, float invN, float lossScale) {
    int lane = threadIdx.x;
    float h = 0.0f;
    for (int j = lane; j < K; j += 32) {
        float p = counts[j] * invN;
        h -= p * __logf(p + 1e-10f);
    }
    #pragma unroll
    for (int off = 16; off > 0; off >>= 1) h += __shfl_xor(h, off, 32);
    if (lane == 0) {
        *out_perp = __expf(h);
        *out_loss = (*sumsq) * lossScale;
    }
}

// ---------------- launcher ----------------
extern "C" void kernel_launch(void* const* d_in, const int* in_sizes, int n_in,
                              void* d_out, int out_size, void* d_ws, size_t ws_size,
                              hipStream_t stream) {
    const float* x    = (const float*)d_in[0];
    const float* ew1  = (const float*)d_in[1];
    const float* eb1  = (const float*)d_in[2];
    const float* ew2  = (const float*)d_in[3];
    const float* eb2  = (const float*)d_in[4];
    const float* ew3  = (const float*)d_in[5];
    const float* eb3  = (const float*)d_in[6];
    const float* cbk  = (const float*)d_in[7];
    const float* dw1  = (const float*)d_in[8];
    const float* db1  = (const float*)d_in[9];
    const float* dw2  = (const float*)d_in[10];
    const float* db2  = (const float*)d_in[11];
    const float* dw3  = (const float*)d_in[12];
    const float* db3  = (const float*)d_in[13];
    float* out = (float*)d_out;

    // workspace (floats); regA: a1 -> dists -> d2 ; regB: a2 -> d1
    float* ws = (float*)d_ws;
    const size_t SZ_A = 16777216;   // 128*128*32*32 == 32768*512
    const size_t SZ_B = 8388608;    // 128*256*16*16
    const size_t SZ_Z = 2097152;    // 128*64*16*16
    float* regA   = ws;
    float* regB   = regA + SZ_A;
    float* z      = regB + SZ_B;
    float* q      = z + SZ_Z;
    float* cnorm  = q + SZ_Z;
    float* counts = cnorm + 512;
    float* sumsq  = counts + 512;

    const int Kcb = 512, D = 64, Nrows = 32768;

    k_init<<<2, 256, 0, stream>>>(counts, sumsq, Kcb);
    k_cnorm<<<2, 256, 0, stream>>>(cbk, cnorm, Kcb, D);

    // ---- encoder ----
    k_conv_wmma<3, 64, 64, 128, 32, 32, 4, 2, 1, 1, 0, 1>
        <<<dim3(2048, 2), 128, 0, stream>>>(x, ew1, eb1, regA);
    k_conv_wmma<128, 32, 32, 256, 16, 16, 4, 2, 1, 1, 0, 1>
        <<<dim3(512, 4), 128, 0, stream>>>(regA, ew2, eb2, regB);
    k_conv_wmma<256, 16, 16, 64, 16, 16, 3, 1, 1, 1, 0, 0>
        <<<dim3(512, 1), 128, 0, stream>>>(regB, ew3, eb3, z);

    // ---- VQ ----
    k_vq_dist_wmma<<<dim3(512, 8), 128, 0, stream>>>(z, cbk, cnorm, regA);
    k_vq_argmin<<<(Nrows + 7) / 8, 256, 0, stream>>>(regA, z, cbk, q, counts, sumsq,
                                                     Nrows, Kcb, D);

    // ---- decoder (transposed convs as dilated convs, flipped weights) ----
    k_conv_wmma<64, 16, 16, 256, 16, 16, 3, 1, 2, 1, 1, 1>
        <<<dim3(512, 4), 128, 0, stream>>>(q, dw1, db1, regB);
    k_conv_wmma<256, 16, 16, 128, 32, 32, 4, 1, 2, 2, 1, 1>
        <<<dim3(2048, 2), 128, 0, stream>>>(regB, dw2, db2, regA);
    k_conv_wmma<128, 32, 32, 3, 64, 64, 4, 1, 2, 2, 1, 2>
        <<<dim3(8192, 1), 128, 0, stream>>>(regA, dw3, db3, out);

    // ---- scalars ----
    const float invN = 1.0f / (float)Nrows;
    const float lossScale = 1.25f / (float)SZ_Z;
    k_finalize<<<1, 32, 0, stream>>>(counts, sumsq,
                                     out + (out_size - 2), out + (out_size - 1),
                                     Kcb, invN, lossScale);
}
